// SimpleGAT_28200755266185
// MI455X (gfx1250) — matrix operand order repaired and checked by reading, not compiled, not run on hardware
//
#include <hip/hip_runtime.h>
#include <hip/hip_bf16.h>

typedef __attribute__((ext_vector_type(16))) _Float16 v16h;
typedef __attribute__((ext_vector_type(8)))  float    v8f;

#define F_IN 128
#define HDIM 16
#define CDIM 8
#define NEG_SLOPE 0.2f

// ---------- monotone float<->int key for atomicMax on floats ----------
__device__ __forceinline__ int fkey(float f) {
    int b = __float_as_int(f);
    return b ^ ((b >> 31) & 0x7fffffff);
}
__device__ __forceinline__ float funkey(int k) {
    int b = k ^ ((k >> 31) & 0x7fffffff);
    return __int_as_float(b);
}

__device__ __forceinline__ void cvt8(v16h& a, int base, float4 p, float4 q) {
    a[base + 0] = (_Float16)p.x; a[base + 1] = (_Float16)p.y;
    a[base + 2] = (_Float16)p.z; a[base + 3] = (_Float16)p.w;
    a[base + 4] = (_Float16)q.x; a[base + 5] = (_Float16)q.y;
    a[base + 6] = (_Float16)q.z; a[base + 7] = (_Float16)q.w;
}

// =====================================================================
// Layer-1 GEMM: h1 = x @ W1 (200k x 128 @ 128 x 16) via WMMA f16,
// fused alpha_s = h1 @ a_src, alpha_d = h1 @ a_dst.
// One wave32 owns a 16-row tile; K=128 -> 4 x wmma_f32_16x16x32_f16.
// B fragments are pre-swizzled in LDS to the per-lane WMMA layout so
// each lane fetches v16h as 2 x ds_load_b128.
// =====================================================================
__global__ __launch_bounds__(256) void gat_gemm1(
    const float* __restrict__ x, const float* __restrict__ W1,
    const float* __restrict__ a_src, const float* __restrict__ a_dst,
    float* __restrict__ h, float* __restrict__ as_, float* __restrict__ ad_,
    int n)
{
    // bfrag[c][lane][e] = W1[(c*32 + (e>>3)*16 + (lane>>4)*8 + (e&7))][lane&15]
    __shared__ __align__(32) _Float16 bfrag[4 * 32 * 16];
    for (int i = threadIdx.x; i < 4 * 32 * 16; i += blockDim.x) {
        int c    = i >> 9;
        int lane = (i >> 4) & 31;
        int e    = i & 15;
        int k    = c * 32 + (e >> 3) * 16 + (lane >> 4) * 8 + (e & 7);
        bfrag[i] = (_Float16)W1[k * HDIM + (lane & 15)];
    }
    __syncthreads();

    const int lane = threadIdx.x & 31;
    const int wv   = threadIdx.x >> 5;
    const int tile = blockIdx.x * 8 + wv;
    const int row0 = tile * 16;
    const int col  = lane & 15;
    const int hf   = lane >> 4;          // half-wave id
    const int r    = row0 + col;         // A row owned by this lane
    const bool rin = (r < n);
    const float* xr = x + (size_t)(rin ? r : 0) * F_IN;  // clamped: always safe

    v8f acc = {};
#pragma unroll
    for (int c = 0; c < 4; ++c) {
        // A layout: a[0..7] = x[row][k0..k0+7], a[8..15] = x[row][k0+16..k0+23]
        const int k0 = c * 32 + hf * 8;
        float4 p0 = *(const float4*)(xr + k0);
        float4 p1 = *(const float4*)(xr + k0 + 4);
        float4 q0 = *(const float4*)(xr + k0 + 16);
        float4 q1 = *(const float4*)(xr + k0 + 20);
        v16h a;
        cvt8(a, 0, p0, p1);
        cvt8(a, 8, q0, q1);
        v16h b = *(const v16h*)(bfrag + ((c * 32 + lane) << 4));
        acc = __builtin_amdgcn_wmma_f32_16x16x32_f16(
            false, a, false, b, (short)0, acc, false, false);
    }

    const float asc = a_src[col];
    const float adc = a_dst[col];
    const bool full = (row0 + 16 <= n);
#pragma unroll
    for (int g = 0; g < 8; ++g) {
        // C/D layout: VGPR g, lanes 0-15 -> M=g ; lanes 16-31 -> M=g+8
        int rowM = row0 + g + hf * 8;
        float hv = acc[g];
        if (full || rowM < n) h[(size_t)rowM * HDIM + col] = hv;
        float ps = hv * asc;
        float pd = hv * adc;
#pragma unroll
        for (int off = 1; off < 16; off <<= 1) {
            ps += __shfl_xor(ps, off, 32);
            pd += __shfl_xor(pd, off, 32);
        }
        if (col == 0 && (full || rowM < n)) { as_[rowM] = ps; ad_[rowM] = pd; }
    }
}

// =====================================================================
// Layer-2 GEMM: h2 = relu(agg1 + b1) @ W2  (K=16 zero-padded to 32,
// N=8 zero-padded to 16) -> single WMMA per 16-row tile; fused alphas.
// =====================================================================
__global__ __launch_bounds__(256) void gat_gemm2(
    const float* __restrict__ agg1, const float* __restrict__ b1,
    const float* __restrict__ W2, const float* __restrict__ a_src,
    const float* __restrict__ a_dst, float* __restrict__ h2,
    float* __restrict__ as_, float* __restrict__ ad_, int n)
{
    // bfrag[lane][e] = W2pad[(e>>3)*16 + (lane>>4)*8 + (e&7)][lane&15]
    __shared__ __align__(32) _Float16 bfrag[32 * 16];
    for (int i = threadIdx.x; i < 32 * 16; i += blockDim.x) {
        int lane = i >> 4;
        int e    = i & 15;
        int k    = (e >> 3) * 16 + (lane >> 4) * 8 + (e & 7);
        int cc   = lane & 15;
        bfrag[i] = (k < HDIM && cc < CDIM) ? (_Float16)W2[k * CDIM + cc]
                                           : (_Float16)0.f;
    }
    __syncthreads();

    const int lane = threadIdx.x & 31;
    const int wv   = threadIdx.x >> 5;
    const int tile = blockIdx.x * 8 + wv;
    const int row0 = tile * 16;
    const int col  = lane & 15;
    const int hf   = lane >> 4;
    const int r    = row0 + col;
    const bool rin = (r < n);
    const float* ar = agg1 + (size_t)(rin ? r : 0) * HDIM;  // clamped

    const int k0 = hf * 8;
    float4 h0 = *(const float4*)(ar + k0);
    float4 h1 = *(const float4*)(ar + k0 + 4);
    float4 g0 = *(const float4*)(b1 + k0);
    float4 g1 = *(const float4*)(b1 + k0 + 4);
    h0.x = fmaxf(h0.x + g0.x, 0.f); h0.y = fmaxf(h0.y + g0.y, 0.f);
    h0.z = fmaxf(h0.z + g0.z, 0.f); h0.w = fmaxf(h0.w + g0.w, 0.f);
    h1.x = fmaxf(h1.x + g1.x, 0.f); h1.y = fmaxf(h1.y + g1.y, 0.f);
    h1.z = fmaxf(h1.z + g1.z, 0.f); h1.w = fmaxf(h1.w + g1.w, 0.f);

    v16h a = {};                 // K in [16,32) stays zero (padding)
    cvt8(a, 0, h0, h1);
    v16h b = *(const v16h*)(bfrag + (lane << 4));

    v8f acc = {};
    acc = __builtin_amdgcn_wmma_f32_16x16x32_f16(
        false, a, false, b, (short)0, acc, false, false);

    const float asc = (col < CDIM) ? a_src[col] : 0.f;
    const float adc = (col < CDIM) ? a_dst[col] : 0.f;
    const bool full = (row0 + 16 <= n);
#pragma unroll
    for (int g = 0; g < 8; ++g) {
        int rowM = row0 + g + hf * 8;
        float hv = acc[g];
        if (col < CDIM && (full || rowM < n)) h2[(size_t)rowM * CDIM + col] = hv;
        float ps = hv * asc;
        float pd = hv * adc;
#pragma unroll
        for (int off = 1; off < 16; off <<= 1) {
            ps += __shfl_xor(ps, off, 32);
            pd += __shfl_xor(pd, off, 32);
        }
        if (col == 0 && (full || rowM < n)) { as_[rowM] = ps; ad_[rowM] = pd; }
    }
}

// =====================================================================
// Edge kernels (shared by both layers). idx >= E encodes self loops.
// Per-node state is L2-resident (192MB L2), so gathers/atomics hit L2.
// =====================================================================
__device__ __forceinline__ void edge_pair(const int* __restrict__ ei, int E,
                                          int i, int& s, int& d) {
    if (i < E) { s = ei[i]; d = ei[E + i]; }
    else       { s = d = i - E; }
}

__global__ void edge_max_k(const int* __restrict__ ei,
                           const float* __restrict__ as_,
                           const float* __restrict__ ad_,
                           int* __restrict__ mkey, int E, int n)
{
    int i = blockIdx.x * blockDim.x + threadIdx.x;
    int total = E + n;
    if (i >= total) return;
    __builtin_prefetch(ei + i + 8192, 0, 1);          // global_prefetch_b8
    int s, d; edge_pair(ei, E, i, s, d);
    float e = as_[s] + ad_[d];
    e = e > 0.f ? e : NEG_SLOPE * e;                  // leaky relu
    atomicMax(&mkey[d], fkey(e));                     // global_atomic_max_i32
}

__global__ void edge_denom_k(const int* __restrict__ ei,
                             const float* __restrict__ as_,
                             const float* __restrict__ ad_,
                             const int* __restrict__ mkey,
                             float* __restrict__ den, int E, int n)
{
    int i = blockIdx.x * blockDim.x + threadIdx.x;
    int total = E + n;
    if (i >= total) return;
    int s, d; edge_pair(ei, E, i, s, d);
    float e = as_[s] + ad_[d];
    e = e > 0.f ? e : NEG_SLOPE * e;
    float w = __expf(e - funkey(mkey[d]));
    unsafeAtomicAdd(&den[d], w);                      // global_atomic_add_f32
}

template <int HH>
__global__ void edge_agg_k(const int* __restrict__ ei,
                           const float* __restrict__ as_,
                           const float* __restrict__ ad_,
                           const int* __restrict__ mkey,
                           const float* __restrict__ den,
                           const float* __restrict__ h,
                           float* __restrict__ agg, int E, int n)
{
    int i = blockIdx.x * blockDim.x + threadIdx.x;
    int total = E + n;
    if (i >= total) return;
    int s, d; edge_pair(ei, E, i, s, d);
    float e = as_[s] + ad_[d];
    e = e > 0.f ? e : NEG_SLOPE * e;
    float w = __expf(e - funkey(mkey[d])) / den[d];
    const float* hs = h + (size_t)s * HH;
    float* od = agg + (size_t)d * HH;
#pragma unroll
    for (int c = 0; c < HH; c += 4) {
        float4 hv = *(const float4*)(hs + c);
        unsafeAtomicAdd(od + c + 0, hv.x * w);
        unsafeAtomicAdd(od + c + 1, hv.y * w);
        unsafeAtomicAdd(od + c + 2, hv.z * w);
        unsafeAtomicAdd(od + c + 3, hv.w * w);
    }
}

// ---------- init: agg buffers to 0, max keys to INT_MIN, denoms to 0 ----
__global__ void init_ws_k(float* __restrict__ agg1, float* __restrict__ agg2,
                          int* __restrict__ mk1, int* __restrict__ mk2,
                          float* __restrict__ den1, float* __restrict__ den2,
                          int n)
{
    int i = blockIdx.x * blockDim.x + threadIdx.x;
    if (i < n * HDIM) agg1[i] = 0.f;
    if (i < n * CDIM) agg2[i] = 0.f;
    if (i < n) {
        mk1[i] = (int)0x80000000; mk2[i] = (int)0x80000000;
        den1[i] = 0.f; den2[i] = 0.f;
    }
}

// ---------- final: out = log_softmax(agg2 + b2) -------------------------
__global__ void final_lsm_k(const float* __restrict__ agg2,
                            const float* __restrict__ b2,
                            float* __restrict__ out, int n)
{
    int i = blockIdx.x * blockDim.x + threadIdx.x;
    if (i >= n) return;
    float v[CDIM];
    float m = -3.0e38f;
#pragma unroll
    for (int c = 0; c < CDIM; ++c) {
        v[c] = agg2[(size_t)i * CDIM + c] + b2[c];
        m = fmaxf(m, v[c]);
    }
    float s = 0.f;
#pragma unroll
    for (int c = 0; c < CDIM; ++c) s += __expf(v[c] - m);
    float l = __logf(s) + m;
#pragma unroll
    for (int c = 0; c < CDIM; ++c) out[(size_t)i * CDIM + c] = v[c] - l;
}

// =====================================================================
extern "C" void kernel_launch(void* const* d_in, const int* in_sizes, int n_in,
                              void* d_out, int out_size, void* d_ws, size_t ws_size,
                              hipStream_t stream)
{
    const float* x    = (const float*)d_in[0];
    const int*   ei   = (const int*)  d_in[1];
    const float* W1   = (const float*)d_in[2];
    const float* as1w = (const float*)d_in[3];
    const float* ad1w = (const float*)d_in[4];
    const float* b1   = (const float*)d_in[5];
    const float* W2   = (const float*)d_in[6];
    const float* as2w = (const float*)d_in[7];
    const float* ad2w = (const float*)d_in[8];
    const float* b2   = (const float*)d_in[9];
    float* out = (float*)d_out;

    const int n = in_sizes[0] / F_IN;
    const int E = in_sizes[1] / 2;

    // carve workspace (~45 MB, L2-resident working set)
    float* p = (float*)d_ws;
    float* h1   = p; p += (size_t)n * HDIM;
    float* as1  = p; p += n;
    float* ad1  = p; p += n;
    int*   mk1  = (int*)p; p += n;
    float* den1 = p; p += n;
    float* agg1 = p; p += (size_t)n * HDIM;
    float* h2   = p; p += (size_t)n * CDIM;
    float* as2  = p; p += n;
    float* ad2  = p; p += n;
    int*   mk2  = (int*)p; p += n;
    float* den2 = p; p += n;
    float* agg2 = p; p += (size_t)n * CDIM;

    const int nt   = 256;
    const int gInit = (n * HDIM + nt - 1) / nt;
    const int tiles = (n + 15) / 16;
    const int gGemm = (tiles + 7) / 8;          // 8 waves/block, 1 tile/wave
    const int gEdge = (E + n + nt - 1) / nt;
    const int gNode = (n + nt - 1) / nt;

    init_ws_k<<<gInit, nt, 0, stream>>>(agg1, agg2, mk1, mk2, den1, den2, n);

    // ---- layer 1 ----
    gat_gemm1<<<gGemm, nt, 0, stream>>>(x, W1, as1w, ad1w, h1, as1, ad1, n);
    edge_max_k  <<<gEdge, nt, 0, stream>>>(ei, as1, ad1, mk1, E, n);
    edge_denom_k<<<gEdge, nt, 0, stream>>>(ei, as1, ad1, mk1, den1, E, n);
    edge_agg_k<HDIM><<<gEdge, nt, 0, stream>>>(ei, as1, ad1, mk1, den1, h1, agg1, E, n);

    // ---- layer 2 (ReLU + b1 fused into GEMM2 A-operand) ----
    gat_gemm2<<<gGemm, nt, 0, stream>>>(agg1, b1, W2, as2w, ad2w, h2, as2, ad2, n);
    edge_max_k  <<<gEdge, nt, 0, stream>>>(ei, as2, ad2, mk2, E, n);
    edge_denom_k<<<gEdge, nt, 0, stream>>>(ei, as2, ad2, mk2, den2, E, n);
    edge_agg_k<CDIM><<<gEdge, nt, 0, stream>>>(ei, as2, ad2, mk2, den2, h2, agg2, E, n);

    // ---- log_softmax(agg2 + b2) ----
    final_lsm_k<<<gNode, nt, 0, stream>>>(agg2, b2, out, n);
    (void)ws_size; (void)n_in; (void)out_size;
}
